// DCNN2d_69896297775482
// MI455X (gfx1250) — matrix operand order repaired and checked by reading, not compiled, not run on hardware
//
#include <hip/hip_runtime.h>

// ---------------------------------------------------------------------------
// DCT-I spectral operator (FNO-like) for MI455X / gfx1250.
// All large GEMMs go through v_wmma_f32_16x16x32_f16 (wave32 WMMA).
// Residual stream h kept f32 in workspace; operands converted to f16 on load.
// ---------------------------------------------------------------------------

typedef __attribute__((ext_vector_type(16))) _Float16 v16h;
typedef __attribute__((ext_vector_type(8)))  float    v8f;

#define DEV static __device__ __forceinline__

DEV v8f wmma16(v16h a, v16h b, v8f c) {
  // (neg_a, A, neg_b, B, c_mod, C, reuse_a, reuse_b)
  return __builtin_amdgcn_wmma_f32_16x16x32_f16(false, a, false, b, (short)0, c,
                                                false, false);
}

// A fragment (16x32, f16 source, row-major [m][k]).
// lane L: m = L&15, kb = (L>>4)*8 ; j<8 -> k = k0+kb+j ; j>=8 -> k = k0+16+kb+(j-8)
DEV v16h ldA_h(const _Float16* __restrict__ A, int lda, int row, int k0, int lane) {
  const _Float16* p = A + (size_t)row * lda + k0 + ((lane >> 4) << 3);
  v16h a;
#pragma unroll
  for (int t = 0; t < 8; ++t) a[t] = p[t];
#pragma unroll
  for (int t = 0; t < 8; ++t) a[8 + t] = p[16 + t];
  return a;
}

// A fragment from f32 source (converted). Buffers are zero-padded, no guards.
DEV v16h ldA_f32(const float* __restrict__ A, int lda, int row, int k0, int lane) {
  const float* p = A + (size_t)row * lda + k0 + ((lane >> 4) << 3);
  v16h a;
#pragma unroll
  for (int t = 0; t < 8; ++t) a[t] = (_Float16)p[t];
#pragma unroll
  for (int t = 0; t < 8; ++t) a[8 + t] = (_Float16)p[16 + t];
  return a;
}

// B fragment (32x16, row-major [k][n]): lane L: n = n0+(L&15), kb = (L>>4)*8 ; j<8 -> k = k0+kb+j ; j>=8 -> k = k0+16+kb+(j-8)
DEV v16h ldB_h(const _Float16* __restrict__ B, int ldb, int k0, int n0, int lane) {
  const _Float16* p = B + (size_t)(k0 + ((lane >> 4) << 3)) * ldb + n0 + (lane & 15);
  v16h b;
#pragma unroll
  for (int t = 0; t < 8; ++t) b[t] = p[(size_t)t * ldb];
#pragma unroll
  for (int t = 0; t < 8; ++t) b[8 + t] = p[(size_t)(16 + t) * ldb];
  return b;
}

DEV v16h ldB_f32(const float* __restrict__ B, int ldb, int k0, int n0, int lane) {
  const float* p = B + (size_t)(k0 + ((lane >> 4) << 3)) * ldb + n0 + (lane & 15);
  v16h b;
#pragma unroll
  for (int t = 0; t < 8; ++t) b[t] = (_Float16)p[(size_t)t * ldb];
#pragma unroll
  for (int t = 0; t < 8; ++t) b[8 + t] = (_Float16)p[(size_t)(16 + t) * ldb];
  return b;
}

// D store: VGPR v -> row row0 + v + (lane>>4)*8, col col0 + (lane&15)
DEV void stD(float* __restrict__ D, int ldd, int row0, int nrows, int col0,
             const v8f& acc, int lane) {
  int col = col0 + (lane & 15);
  int rb = row0 + ((lane >> 4) << 3);
#pragma unroll
  for (int v = 0; v < 8; ++v) {
    int r = rb + v;
    if (r < nrows) D[(size_t)r * ldd + col] = acc[v];
  }
}

// ---------------------------------------------------------------------------
// Table builder: cosine bases (f16, zero padded) + fc1^T.
// Atab (48x256): A[r][x] = w(x) cos(pi k_r x / 255), k_r = r<20? r : 216+r
// BmT  (256x32): B[y][m] = w(y) cos(pi m y / 255)            (cols >=20 zero)
// DyT  (32x256): D[m][y] = w(m) cos(pi y m / 255)            (rows >=20 zero)
// DxA  (256x64): D[x][r] = w(k_r) cos(pi x k_r / 255)        (cols >=40 zero)
// fc1T (128x64): fc1_w^T
// ---------------------------------------------------------------------------
__global__ void k_init(_Float16* __restrict__ Atab, _Float16* __restrict__ BmT,
                       _Float16* __restrict__ DyT, _Float16* __restrict__ DxA,
                       _Float16* __restrict__ fc1T, const float* __restrict__ fc1w) {
  int i = blockIdx.x * 256 + threadIdx.x;  // 16384 threads
  if (i < 48 * 256) {
    int r = i >> 8, x = i & 255;
    float v = 0.f;
    if (r < 40) {
      int k = (r < 20) ? r : (216 + r);
      float w = (x == 0 || x == 255) ? 1.f : 2.f;
      v = w * cospif((float)(k * x) * (1.f / 255.f));
    }
    Atab[i] = (_Float16)v;
  }
  if (i < 256 * 32) {
    int y = i >> 5, m = i & 31;
    float v = 0.f;
    if (m < 20) {
      float w = (y == 0 || y == 255) ? 1.f : 2.f;
      v = w * cospif((float)(m * y) * (1.f / 255.f));
    }
    BmT[i] = (_Float16)v;
  }
  if (i < 32 * 256) {
    int m = i >> 8, y = i & 255;
    float v = 0.f;
    if (m < 20) {
      float w = (m == 0) ? 1.f : 2.f;
      v = w * cospif((float)(m * y) * (1.f / 255.f));
    }
    DyT[i] = (_Float16)v;
  }
  if (i < 256 * 64) {
    int x = i >> 6, r = i & 63;
    float v = 0.f;
    if (r < 40) {
      int k = (r < 20) ? r : (216 + r);
      float w = (k == 0 || k == 255) ? 1.f : 2.f;
      v = w * cospif((float)(k * x) * (1.f / 255.f));
    }
    DxA[i] = (_Float16)v;
  }
  if (i < 128 * 64) {
    int j = i >> 6, c = i & 63;
    fc1T[i] = (_Float16)fc1w[c * 128 + j];
  }
}

// fc0: h[b,c,s] = bias[c] + sum_d x[b,s,d] * w[d,c]   (K=3 -> plain VALU)
__global__ void k_fc0(const float* __restrict__ x, const float* __restrict__ w,
                      const float* __restrict__ bias, float* __restrict__ h) {
  int idx = blockIdx.x * 256 + threadIdx.x;  // (b*64+c)*65536 + s
  int s = idx & 65535;
  int c = (idx >> 16) & 63;
  int b = idx >> 22;
  const float* xp = x + ((size_t)b * 65536 + s) * 3;
  h[idx] = bias[c] + xp[0] * w[c] + xp[1] * w[64 + c] + xp[2] * w[128 + c];
}

// forward DCT over X: tmp1[bc][r][y] = sum_x Atab[r][x] h[bc][x][y]
// per wave: one 16x16 tile; M=r (3 tiles, 40 valid), N=y (16), K=x (8x32)
__global__ void k_dctx(const float* __restrict__ h, const _Float16* __restrict__ Atab,
                       float* __restrict__ tmp1) {
  int wv = blockIdx.x * 8 + (threadIdx.x >> 5);
  int lane = threadIdx.x & 31;
  int nt = wv & 15, mt = (wv >> 4) % 3, bc = wv / 48;
  const float* hb = h + (size_t)bc * 65536;
  int arow = mt * 16 + (lane & 15);
  v8f acc = {};
#pragma unroll
  for (int k0 = 0; k0 < 256; k0 += 32) {
    v16h a = ldA_h(Atab, 256, arow, k0, lane);
    v16h b = ldB_f32(hb, 256, k0, nt * 16, lane);
    acc = wmma16(a, b, acc);
  }
  stD(tmp1 + (size_t)bc * 48 * 256, 256, mt * 16, 40, nt * 16, acc, lane);
}

// forward DCT over Y: modes[bc][r][m] = sum_y tmp1[bc][r][y] * BmT[y][m]
__global__ void k_dcty(const float* __restrict__ tmp1, const _Float16* __restrict__ BmT,
                       float* __restrict__ modes) {
  int wv = blockIdx.x * 8 + (threadIdx.x >> 5);
  int lane = threadIdx.x & 31;
  int nt = wv & 1, mt = (wv >> 1) % 3, bc = wv / 6;
  const float* tb = tmp1 + (size_t)bc * 48 * 256;
  int arow = mt * 16 + (lane & 15);
  v8f acc = {};
#pragma unroll
  for (int k0 = 0; k0 < 256; k0 += 32) {
    v16h a = ldA_f32(tb, 256, arow, k0, lane);  // rows 40..47 are zero pad
    v16h b = ldB_h(BmT, 32, k0, nt * 16, lane);
    acc = wmma16(a, b, acc);
  }
  stD(modes + (size_t)bc * 48 * 32, 32, mt * 16, 40, nt * 16, acc, lane);
}

// per-mode channel mix: modes2[b,o,r,m] = sum_i modes[b,i,r,m] * w[i,o,r',m]
__global__ void k_mix(const float* __restrict__ modes, const float* __restrict__ w1,
                      const float* __restrict__ w2, float* __restrict__ modes2) {
  int idx = blockIdx.x * 256 + threadIdx.x;  // 409600
  int m = idx % 20;
  int r = (idx / 20) % 40;
  int o = (idx / 800) % 64;
  int b = idx / 51200;
  int xr = (r < 20) ? r : (r - 20);
  const float* wp = ((r < 20) ? w1 : w2) + ((size_t)o * 20 + xr) * 20 + m;  // +i*25600
  const float* ip = modes + ((size_t)(b * 64) * 48 + r) * 32 + m;           // +i*1536
  float acc = 0.f;
#pragma unroll 4
  for (int i = 0; i < 64; ++i)
    acc += ip[(size_t)i * 1536] * wp[(size_t)i * 25600];
  modes2[((size_t)(b * 64 + o) * 48 + r) * 32 + m] = acc;
}

// inverse over Y: tmp2[bo][r][y] = sum_m modes2[bo][r][m] * DyT[m][y]  (K=32, 1 chunk)
__global__ void k_invy(const float* __restrict__ modes2, const _Float16* __restrict__ DyT,
                       float* __restrict__ tmp2) {
  int wv = blockIdx.x * 8 + (threadIdx.x >> 5);
  int lane = threadIdx.x & 31;
  int nt = wv & 15, mt = (wv >> 4) % 3, bo = wv / 48;
  v16h a = ldA_f32(modes2 + (size_t)bo * 48 * 32, 32, mt * 16 + (lane & 15), 0, lane);
  v16h b = ldB_h(DyT, 256, 0, nt * 16, lane);
  v8f acc = {};
  acc = wmma16(a, b, acc);
  stD(tmp2 + (size_t)bo * 64 * 256, 256, mt * 16, 40, nt * 16, acc, lane);
}

// inverse over X: x1[bo][x][y] = sum_r DxA[x][r] * tmp2[bo][r][y]  (K=64, 2 chunks)
__global__ void k_invx(const float* __restrict__ tmp2, const _Float16* __restrict__ DxA,
                       float* __restrict__ hout) {
  int wv = blockIdx.x * 8 + (threadIdx.x >> 5);
  int lane = threadIdx.x & 31;
  int nt = wv & 15, mt = (wv >> 4) & 15, bo = wv >> 8;
  const float* tb = tmp2 + (size_t)bo * 64 * 256;
  int arow = mt * 16 + (lane & 15);
  v8f acc = {};
#pragma unroll
  for (int k0 = 0; k0 < 64; k0 += 32) {
    v16h a = ldA_h(DxA, 64, arow, k0, lane);
    v16h b = ldB_f32(tb, 256, k0, nt * 16, lane);  // rows 40..63 zero pad
    acc = wmma16(a, b, acc);
  }
  float* ho = hout + (size_t)bo * 65536;
  int col = nt * 16 + (lane & 15);
  int rb = mt * 16 + ((lane >> 4) << 3);
#pragma unroll
  for (int v = 0; v < 8; ++v) ho[(size_t)(rb + v) * 256 + col] = acc[v];
}

// 1x1 conv + bias + residual(x1) + optional tanh:
// h_next[b,o,s] = act( sum_i W[o,i] h_in[b,i,s] + cb[o] + hout[b,o,s] )
__global__ void k_conv(const float* __restrict__ hin, const float* __restrict__ cw,
                       const float* __restrict__ cb, float* __restrict__ hout, int act) {
  int wv = blockIdx.x * 8 + (threadIdx.x >> 5);
  int lane = threadIdx.x & 31;
  int st = wv & 4095, mt = (wv >> 12) & 3, b = wv >> 14;
  const float* hi = hin + (size_t)b * 64 * 65536;
  int s0 = st * 16;
  int arow = mt * 16 + (lane & 15);
  v8f acc = {};
#pragma unroll
  for (int k0 = 0; k0 < 64; k0 += 32) {
    v16h a = ldA_f32(cw, 64, arow, k0, lane);      // conv_w[o][i]
    v16h bfr = ldB_f32(hi + (size_t)k0 * 65536, 65536, lane, s0, lane - lane);  // placeholder
    // (explicit form below; keep it clear:)
    {
      const float* p = hi + (size_t)(k0 + ((lane >> 4) << 3)) * 65536 + s0 + (lane & 15);
#pragma unroll
      for (int t = 0; t < 8; ++t) bfr[t] = (_Float16)p[(size_t)t * 65536];
#pragma unroll
      for (int t = 0; t < 8; ++t) bfr[8 + t] = (_Float16)p[(size_t)(16 + t) * 65536];
    }
    acc = wmma16(a, bfr, acc);
  }
  int s = s0 + (lane & 15);
  int ob = mt * 16 + ((lane >> 4) << 3);
#pragma unroll
  for (int v = 0; v < 8; ++v) {
    int o = ob + v;
    float* q = hout + (size_t)(b * 64 + o) * 65536 + s;
    float val = acc[v] + cb[o] + *q;
    if (act) val = tanhf(val);
    *q = val;
  }
}

// fused head: out[p] = fc2_b + sum_j fc2_w[j] * tanh(fc1_b[j] + sum_c fc1T[j][c] h[c,p])
__global__ void k_head(const float* __restrict__ hfin, const _Float16* __restrict__ fc1T,
                       const float* __restrict__ fc1b, const float* __restrict__ fc2w,
                       const float* __restrict__ fc2b, float* __restrict__ out) {
  int wv = blockIdx.x * 8 + (threadIdx.x >> 5);
  int lane = threadIdx.x & 31;
  int b = wv >> 12;
  int s0 = (wv & 4095) * 16;
  const float* hb = hfin + (size_t)b * 64 * 65536;
  float partial = 0.f;
#pragma unroll
  for (int tj = 0; tj < 8; ++tj) {
    v8f acc = {};
#pragma unroll
    for (int k0 = 0; k0 < 64; k0 += 32) {
      v16h a = ldA_h(fc1T, 64, tj * 16 + (lane & 15), k0, lane);
      v16h bfr;
      const float* p = hb + (size_t)(k0 + ((lane >> 4) << 3)) * 65536 + s0 + (lane & 15);
#pragma unroll
      for (int t = 0; t < 8; ++t) bfr[t] = (_Float16)p[(size_t)t * 65536];
#pragma unroll
      for (int t = 0; t < 8; ++t) bfr[8 + t] = (_Float16)p[(size_t)(16 + t) * 65536];
      acc = wmma16(a, bfr, acc);
    }
    int jb = tj * 16 + ((lane >> 4) << 3);
#pragma unroll
    for (int v = 0; v < 8; ++v) {
      int j = jb + v;
      partial += tanhf(acc[v] + fc1b[j]) * fc2w[j];
    }
  }
  partial += __shfl_xor(partial, 16);  // wave32: lanes l and l^16 share a pixel
  if (lane < 16) out[(size_t)b * 65536 + s0 + lane] = partial + fc2b[0];
}

// ---------------------------------------------------------------------------
extern "C" void kernel_launch(void* const* d_in, const int* in_sizes, int n_in,
                              void* d_out, int out_size, void* d_ws, size_t ws_size,
                              hipStream_t stream) {
  (void)in_sizes; (void)n_in; (void)out_size; (void)ws_size;
  const float* x     = (const float*)d_in[0];
  const float* fc0w  = (const float*)d_in[1];
  const float* fc0b  = (const float*)d_in[2];
  const float* sp1   = (const float*)d_in[3];
  const float* sp2   = (const float*)d_in[4];
  const float* convw = (const float*)d_in[5];
  const float* convb = (const float*)d_in[6];
  const float* fc1w  = (const float*)d_in[7];
  const float* fc1b  = (const float*)d_in[8];
  const float* fc2w  = (const float*)d_in[9];
  const float* fc2b  = (const float*)d_in[10];

  char* w = (char*)d_ws;
  _Float16* Atab = (_Float16*)(w);             // 48*256
  _Float16* BmT  = (_Float16*)(w + 24576);     // 256*32
  _Float16* DyT  = (_Float16*)(w + 40960);     // 32*256
  _Float16* DxA  = (_Float16*)(w + 57344);     // 256*64
  _Float16* fc1T = (_Float16*)(w + 90112);     // 128*64
  float* hA   = (float*)(w + 106496);          // 8*64*256*256 f32
  float* hB   = hA + 33554432;
  float* tmp1 = hB + 33554432;                 // 512*48*256 (rows 40..47 pad)
  float* tmp2 = tmp1 + (size_t)512 * 48 * 256; // 512*64*256 (rows 40..63 pad)
  float* modesA = tmp2 + (size_t)512 * 64 * 256;   // 512*48*32
  float* modesB = modesA + (size_t)512 * 48 * 32;  // 512*48*32

  // Zero scratch so WMMA operand loads need no guards (pads multiply as 0).
  size_t zbytes = ((size_t)512 * 48 * 256 + (size_t)512 * 64 * 256 +
                   (size_t)2 * 512 * 48 * 32) * sizeof(float);
  hipMemsetAsync(tmp1, 0, zbytes, stream);

  k_init<<<64, 256, 0, stream>>>(Atab, BmT, DyT, DxA, fc1T, fc1w);
  k_fc0<<<131072, 256, 0, stream>>>(x, fc0w, fc0b, hA);

  float* hin = hA;
  float* hout = hB;
  for (int l = 0; l < 3; ++l) {
    k_dctx<<<3072, 256, 0, stream>>>(hin, Atab, tmp1);
    k_dcty<<<384, 256, 0, stream>>>(tmp1, BmT, modesA);
    k_mix<<<1600, 256, 0, stream>>>(modesA, sp1 + (size_t)l * 64 * 64 * 400,
                                    sp2 + (size_t)l * 64 * 64 * 400, modesB);
    k_invy<<<3072, 256, 0, stream>>>(modesB, DyT, tmp2);
    k_invx<<<16384, 256, 0, stream>>>(tmp2, DxA, hout);
    k_conv<<<16384, 256, 0, stream>>>(hin, convw + (size_t)l * 4096,
                                      convb + (size_t)l * 64, hout, (l < 2) ? 1 : 0);
    float* t = hin; hin = hout; hout = t;
  }
  k_head<<<4096, 256, 0, stream>>>(hin, fc1T, fc1b, fc2w, fc2b, (float*)d_out);
}
